// NLBlock_59803124630344
// MI455X (gfx1250) — matrix operand order repaired and checked by reading, not compiled, not run on hardware
//
#include <hip/hip_runtime.h>
#include <hip/hip_bf16.h>

// ---------------- problem constants ----------------
#define B_   4
#define C_   256
#define CI_  128
#define T_   16
#define H_   28
#define W_   28
#define N_   (T_*H_*W_)      // 12544
#define TP_  8
#define HP_  14
#define WP_  14
#define NP_  (TP_*HP_*WP_)   // 1568

typedef __attribute__((ext_vector_type(16))) __bf16 bf16x16;
typedef __attribute__((ext_vector_type(8)))  __bf16 bf16x8;
typedef __attribute__((ext_vector_type(8)))  float  f32x8;

union ABFrag { bf16x16 v; bf16x8 h[2]; };

__device__ __forceinline__ unsigned short f2bf(float f) {
  unsigned int u = __float_as_uint(f);
  u += 0x7FFFu + ((u >> 16) & 1u);          // round-to-nearest-even
  return (unsigned short)(u >> 16);
}
__device__ __forceinline__ float bf2f(unsigned short s) {
  return __uint_as_float(((unsigned int)s) << 16);
}

// ---------------- kernel 0: convert all weights to bf16 ----------------
// layout in ws: [theta_w | phi_w | g_w | z_w], each Ci*C = 32768 elems
__global__ void wconv_kernel(const float* tw, const float* pw, const float* gw,
                             const float* zw, unsigned short* o) {
  int i = blockIdx.x * 256 + threadIdx.x;
  const int n = CI_ * C_;
  if      (i <     n) o[i] = f2bf(tw[i]);
  else if (i < 2 * n) o[i] = f2bf(pw[i - n]);
  else if (i < 3 * n) o[i] = f2bf(gw[i - 2 * n]);
  else if (i < 4 * n) o[i] = f2bf(zw[i - 3 * n]);
}

// ---------------- kernel 1: three projections via WMMA ----------------
// out[n][o] = sum_c x[c][n] * w[o][c].  A = x^T tile (16 x 256 bf16, staged
// transposed in LDS), B = w rows (contiguous over c).  Block = 4 waves,
// each wave owns 2 ci-tiles; 8 K-steps of 32.
__global__ void __launch_bounds__(128)
proj_kernel(const float* __restrict__ x, const unsigned short* __restrict__ wbf,
            unsigned short* __restrict__ theta, unsigned short* __restrict__ phiPre,
            unsigned short* __restrict__ gPre) {
  __shared__ __align__(32) unsigned short xt[16 * C_];   // 8 KB
  const int b   = blockIdx.y;
  const int n0  = blockIdx.x * 16;
  const int tid = threadIdx.x;
  const float* xb = x + (size_t)b * C_ * N_;
  for (int i = tid; i < 16 * C_; i += 128) {
    int j = i & 15, c = i >> 4;
    xt[j * C_ + c] = f2bf(xb[(size_t)c * N_ + n0 + j]);   // transpose into LDS
  }
  __syncthreads();

  const int lane = tid & 31;
  const int wave = tid >> 5;
  const int col  = lane & 15;      // A row / B,D column within tile
  const int hi   = lane >> 4;      // lane half
  const int aOff = hi * 16;        // A-frag K sub-offset (contiguous K, matches B)
  const int bOff = hi * 16;        // B-frag K sub-offset

#pragma unroll
  for (int wsel = 0; wsel < 3; ++wsel) {
    const unsigned short* w = wbf + (size_t)wsel * CI_ * C_;
    unsigned short* dst = (wsel == 0 ? theta : (wsel == 1 ? phiPre : gPre))
                          + (size_t)b * N_ * CI_;
#pragma unroll
    for (int tt = 0; tt < 2; ++tt) {
      const int ci0 = (wave * 2 + tt) * 16;
      f32x8 acc = {};
#pragma unroll
      for (int k0 = 0; k0 < C_; k0 += 32) {
        ABFrag a;
        a.h[0] = *(const bf16x8*)(xt + col * C_ + k0 + aOff);
        a.h[1] = *(const bf16x8*)(xt + col * C_ + k0 + 8 + aOff);
        bf16x16 bf = *(const bf16x16*)(w + (size_t)(ci0 + col) * C_ + k0 + bOff);
        acc = __builtin_amdgcn_wmma_f32_16x16x32_bf16(false, a.v, false, bf,
                                                      (short)0, acc, false, false);
      }
#pragma unroll
      for (int r = 0; r < 8; ++r) {
        int row = n0 + r + hi * 8;
        dst[(size_t)row * CI_ + ci0 + col] = f2bf(acc[r]);
      }
    }
  }
}

// ---------------- kernel 2: 2x2x2 maxpool; phi -> [Np][Ci], g -> [Ci][Np] ----
__global__ void pool_kernel(const unsigned short* __restrict__ phiPre,
                            const unsigned short* __restrict__ gPre,
                            unsigned short* __restrict__ phiP,
                            unsigned short* __restrict__ gPT) {
  int idx = blockIdx.x * 256 + threadIdx.x;
  if (idx >= B_ * NP_ * CI_) return;
  int ci = idx & (CI_ - 1);
  int np = (idx / CI_) % NP_;
  int b  = idx / (CI_ * NP_);
  int tp = np / (HP_ * WP_);
  int rem = np % (HP_ * WP_);
  int hp = rem / WP_, wp = rem % WP_;
  const unsigned short* pb = phiPre + (size_t)b * N_ * CI_;
  const unsigned short* gb = gPre   + (size_t)b * N_ * CI_;
  float pm = -3.4e38f, gm = -3.4e38f;
#pragma unroll
  for (int dt = 0; dt < 2; ++dt)
#pragma unroll
    for (int dh = 0; dh < 2; ++dh)
#pragma unroll
      for (int dw = 0; dw < 2; ++dw) {
        int n = ((2 * tp + dt) * H_ + (2 * hp + dh)) * W_ + (2 * wp + dw);
        pm = fmaxf(pm, bf2f(pb[(size_t)n * CI_ + ci]));
        gm = fmaxf(gm, bf2f(gb[(size_t)n * CI_ + ci]));
      }
  phiP[(size_t)b * NP_ * CI_ + (size_t)np * CI_ + ci] = f2bf(pm);
  gPT [(size_t)b * CI_ * NP_ + (size_t)ci * NP_ + np] = f2bf(gm);
}

// ---------------- kernel 3: flash attention (scores never hit HBM) ---------
// Block = 8 waves / 128 rows.  Per 32-column chunk the block stages the
// phi tile (32x128 bf16, contiguous 8KB) and g tile (128x32 bf16) into LDS
// once, so every wave's B-fragments come from LDS instead of 8x duplicated
// global traffic.  Online softmax keeps the 315MB score matrix out of HBM.
#define AW_ 8
__global__ void __launch_bounds__(256)
attn_kernel(const unsigned short* __restrict__ theta,
            const unsigned short* __restrict__ phiP,
            const unsigned short* __restrict__ gPT,
            unsigned short* __restrict__ yBf) {
  __shared__ __align__(32) unsigned short phiT[32 * CI_];      // 8 KB [col][ci]
  __shared__ __align__(32) unsigned short gT[CI_ * 32];        // 8 KB [ci][col]
  __shared__ __align__(32) unsigned short pbuf[AW_][16 * 32];  // 8 KB
  const int b    = blockIdx.y;
  const int tid  = threadIdx.x;
  const int lane = tid & 31, wave = tid >> 5;
  const int col  = lane & 15, hi = lane >> 4;
  const int aOff = hi * 16, bOff = hi * 16;
  const int n0   = blockIdx.x * (16 * AW_) + wave * 16;

  const unsigned short* thB = theta + (size_t)b * N_ * CI_ + (size_t)(n0 + col) * CI_;
  const unsigned short* phB = phiP  + (size_t)b * NP_ * CI_;
  const unsigned short* gB  = gPT   + (size_t)b * CI_ * NP_;
  unsigned short* pw = pbuf[wave];

  ABFrag aTh[4];
#pragma unroll
  for (int kk = 0; kk < 4; ++kk) {
    aTh[kk].h[0] = *(const bf16x8*)(thB + kk * 32 + aOff);
    aTh[kk].h[1] = *(const bf16x8*)(thB + kk * 32 + 8 + aOff);
  }

  f32x8 zero = {};
  f32x8 yacc[8];
  float rmax[8], rsum[8];
#pragma unroll
  for (int t = 0; t < 8; ++t) yacc[t] = zero;
#pragma unroll
  for (int r = 0; r < 8; ++r) { rmax[r] = -3.4e38f; rsum[r] = 0.f; }

  for (int j0 = 0; j0 < NP_; j0 += 32) {
    __syncthreads();   // previous chunk's LDS fragments fully consumed
    // ---- stage phi tile: contiguous 8KB -> LDS (512 x 16B chunks) ----
    {
      const unsigned short* src = phB + (size_t)j0 * CI_;
#pragma unroll
      for (int c = tid; c < 512; c += 256)
        *(uint4*)(phiT + c * 8) = *(const uint4*)(src + c * 8);
      // ---- stage g tile: 128 rows x 64B -> LDS ----
#pragma unroll
      for (int c = tid; c < 512; c += 256) {
        int ci = c >> 2, sub = c & 3;
        *(uint4*)(gT + ci * 32 + sub * 8) =
            *(const uint4*)(gB + (size_t)ci * NP_ + j0 + sub * 8);
      }
    }
    // prefetch next chunk while this one is computed (global_prefetch_b8)
    if (j0 + 32 < NP_) {
      __builtin_prefetch(phB + (size_t)(j0 + 32) * CI_ + (size_t)tid * 16, 0, 1);
      if (tid < CI_)
        __builtin_prefetch(gB + (size_t)tid * NP_ + j0 + 32, 0, 1);
    }
    __syncthreads();

    f32x8 s0 = zero, s1 = zero;
#pragma unroll
    for (int kk = 0; kk < 4; ++kk) {
      bf16x16 b0 = *(const bf16x16*)(phiT + (size_t)col * CI_ + kk * 32 + bOff);
      bf16x16 b1 = *(const bf16x16*)(phiT + (size_t)(col + 16) * CI_ + kk * 32 + bOff);
      s0 = __builtin_amdgcn_wmma_f32_16x16x32_bf16(false, aTh[kk].v, false, b0,
                                                   (short)0, s0, false, false);
      s1 = __builtin_amdgcn_wmma_f32_16x16x32_bf16(false, aTh[kk].v, false, b1,
                                                   (short)0, s1, false, false);
    }
    // online softmax per row (lane half hi=0 -> rows 0..7, hi=1 -> rows 8..15)
#pragma unroll
    for (int r = 0; r < 8; ++r) {
      float mr = fmaxf(s0[r], s1[r]);
#pragma unroll
      for (int d = 8; d >= 1; d >>= 1) mr = fmaxf(mr, __shfl_xor(mr, d, 32));
      float nm    = fmaxf(rmax[r], mr);
      float scale = __expf(rmax[r] - nm);
      rmax[r] = nm;
      float p0 = __expf(s0[r] - nm);
      float p1 = __expf(s1[r] - nm);
      float rs = p0 + p1;
#pragma unroll
      for (int d = 8; d >= 1; d >>= 1) rs += __shfl_xor(rs, d, 32);
      rsum[r] = rsum[r] * scale + rs;
#pragma unroll
      for (int t = 0; t < 8; ++t) yacc[t][r] *= scale;
      int prow = r + hi * 8;
      pw[prow * 32 + col]      = f2bf(p0);   // D-layout -> LDS (row-major P)
      pw[prow * 32 + 16 + col] = f2bf(p1);
    }
    __builtin_amdgcn_wave_barrier();   // same-wave LDS is in-order; stop reordering
    ABFrag aP;
    aP.h[0] = *(const bf16x8*)(pw + col * 32 + aOff);
    aP.h[1] = *(const bf16x8*)(pw + col * 32 + 8 + aOff);
    __builtin_amdgcn_wave_barrier();
#pragma unroll
    for (int t = 0; t < 8; ++t) {
      bf16x16 gb = *(const bf16x16*)(gT + (size_t)(t * 16 + col) * 32 + bOff);
      yacc[t] = __builtin_amdgcn_wmma_f32_16x16x32_bf16(false, aP.v, false, gb,
                                                        (short)0, yacc[t], false, false);
    }
  }

  unsigned short* yOut = yBf + (size_t)b * N_ * CI_;
#pragma unroll
  for (int r = 0; r < 8; ++r) {
    float inv = 1.0f / rsum[r];
    int row = n0 + r + hi * 8;
#pragma unroll
    for (int t = 0; t < 8; ++t)
      yOut[(size_t)row * CI_ + t * 16 + col] = f2bf(yacc[t][r] * inv);
  }
}

// ---------------- kernel 4: z = y x z_w^T via WMMA, z stored [B][N][C] -----
__global__ void __launch_bounds__(128)
zproj_kernel(const unsigned short* __restrict__ yBf,
             const unsigned short* __restrict__ zwBf,
             float* __restrict__ z) {
  const int b   = blockIdx.y;
  const int n0  = blockIdx.x * 16;
  const int tid = threadIdx.x, lane = tid & 31, wave = tid >> 5;
  const int col = lane & 15, hi = lane >> 4;
  const int aOff = hi * 16, bOff = hi * 16;

  const unsigned short* yB = yBf + (size_t)b * N_ * CI_ + (size_t)(n0 + col) * CI_;
  ABFrag aY[4];
#pragma unroll
  for (int kk = 0; kk < 4; ++kk) {
    aY[kk].h[0] = *(const bf16x8*)(yB + kk * 32 + aOff);
    aY[kk].h[1] = *(const bf16x8*)(yB + kk * 32 + 8 + aOff);
  }
  f32x8 zero = {};
#pragma unroll
  for (int t = 0; t < 4; ++t) {
    const int co = (wave * 4 + t) * 16 + col;
    f32x8 acc = zero;
#pragma unroll
    for (int kk = 0; kk < 4; ++kk) {
      bf16x16 bw = *(const bf16x16*)(zwBf + (size_t)co * CI_ + kk * 32 + bOff);
      acc = __builtin_amdgcn_wmma_f32_16x16x32_bf16(false, aY[kk].v, false, bw,
                                                    (short)0, acc, false, false);
    }
#pragma unroll
    for (int r = 0; r < 8; ++r) {
      int row = n0 + r + hi * 8;
      z[((size_t)b * N_ + row) * C_ + co] = acc[r];
    }
  }
}

// ---------------- kernel 5: deterministic per-channel BN stats -------------
__global__ void __launch_bounds__(256)
bnstats_kernel(const float* __restrict__ z, float* __restrict__ sums) {
  __shared__ float sbuf[256], qbuf[256];
  const int c = blockIdx.x;
  float s = 0.f, q = 0.f;
  for (int i = threadIdx.x; i < B_ * N_; i += 256) {
    float v = z[(size_t)i * C_ + c];   // z fits in 192MB L2 -> L2-resident reads
    s += v; q += v * v;
  }
  sbuf[threadIdx.x] = s; qbuf[threadIdx.x] = q;
  __syncthreads();
  for (int st = 128; st > 0; st >>= 1) {
    if (threadIdx.x < st) {
      sbuf[threadIdx.x] += sbuf[threadIdx.x + st];
      qbuf[threadIdx.x] += qbuf[threadIdx.x + st];
    }
    __syncthreads();
  }
  if (threadIdx.x == 0) { sums[c] = sbuf[0]; sums[C_ + c] = qbuf[0]; }
}

// ---------------- kernel 6: BN normalize + residual ------------------------
__global__ void bn_kernel(const float* __restrict__ x, const float* __restrict__ z,
                          const float* __restrict__ sums,
                          const float* __restrict__ gamma, const float* __restrict__ beta,
                          float* __restrict__ out) {
  size_t idx = (size_t)blockIdx.x * 256 + threadIdx.x;
  const size_t total = (size_t)B_ * C_ * N_;
  if (idx >= total) return;
  int n = (int)(idx % N_);
  int c = (int)((idx / N_) % C_);
  int b = (int)(idx / ((size_t)C_ * N_));
  const float cnt = (float)(B_ * N_);
  float mean = sums[c] / cnt;
  float var  = sums[C_ + c] / cnt - mean * mean;
  float inv  = rsqrtf(var + 1e-5f);
  float zv   = z[((size_t)b * N_ + n) * C_ + c];
  out[idx] = x[idx] + (zv - mean) * inv * gamma[c] + beta[c];
}

// ---------------- host-side orchestration ----------------------------------
extern "C" void kernel_launch(void* const* d_in, const int* in_sizes, int n_in,
                              void* d_out, int out_size, void* d_ws, size_t ws_size,
                              hipStream_t stream) {
  const float* x     = (const float*)d_in[0];
  const float* tw    = (const float*)d_in[1];
  const float* pw    = (const float*)d_in[2];
  const float* gw    = (const float*)d_in[3];
  const float* zw    = (const float*)d_in[4];
  const float* gamma = (const float*)d_in[5];
  const float* beta  = (const float*)d_in[6];
  float* out = (float*)d_out;

  char* ws = (char*)d_ws;
  constexpr size_t WB        = (size_t)CI_ * C_ * 2;          // 64 KB per weight
  constexpr size_t THETA_B   = (size_t)B_ * N_  * CI_ * 2;    // 12.8 MB
  constexpr size_t PHIP_B    = (size_t)B_ * NP_ * CI_ * 2;    // 1.6 MB
  constexpr size_t Z_B       = (size_t)B_ * N_  * C_  * 4;    // 51.4 MB

  unsigned short* wbf    = (unsigned short*)(ws);             // theta|phi|g
  unsigned short* zwBf   = (unsigned short*)(ws + 3 * WB);
  size_t off = 4 * WB;
  unsigned short* theta  = (unsigned short*)(ws + off); off += THETA_B;
  unsigned short* phiPre = (unsigned short*)(ws + off); off += THETA_B;
  unsigned short* gPre   = (unsigned short*)(ws + off); off += THETA_B;
  unsigned short* phiP   = (unsigned short*)(ws + off); off += PHIP_B;
  unsigned short* gPT    = (unsigned short*)(ws + off); off += PHIP_B;
  unsigned short* yBf    = (unsigned short*)(ws + off); off += THETA_B;
  float*          zbuf   = (float*)(ws + off);          off += Z_B;
  float*          sums   = (float*)(ws + off);          // 2*C floats

  hipLaunchKernelGGL(wconv_kernel, dim3(512), dim3(256), 0, stream,
                     tw, pw, gw, zw, (unsigned short*)ws);
  hipLaunchKernelGGL(proj_kernel, dim3(N_ / 16, B_), dim3(128), 0, stream,
                     x, wbf, theta, phiPre, gPre);
  hipLaunchKernelGGL(pool_kernel, dim3((B_ * NP_ * CI_ + 255) / 256), dim3(256), 0, stream,
                     phiPre, gPre, phiP, gPT);
  hipLaunchKernelGGL(attn_kernel, dim3(N_ / (16 * AW_), B_), dim3(256), 0, stream,
                     theta, phiP, gPT, yBf);
  hipLaunchKernelGGL(zproj_kernel, dim3(N_ / 16, B_), dim3(128), 0, stream,
                     yBf, zwBf, zbuf);
  hipLaunchKernelGGL(bnstats_kernel, dim3(C_), dim3(256), 0, stream, zbuf, sums);
  const size_t total = (size_t)B_ * C_ * N_;
  hipLaunchKernelGGL(bn_kernel, dim3((unsigned)((total + 255) / 256)), dim3(256), 0, stream,
                     x, zbuf, sums, gamma, beta, out);
}
